// MambaBlock_48988396978565
// MI455X (gfx1250) — matrix operand order repaired and checked by reading, not compiled, not run on hardware
//
#include <hip/hip_runtime.h>
#include <hip/hip_bf16.h>
#include <math.h>

// ---------------- problem constants ----------------
#define BDIM   2
#define LSEQ   1024
#define DMODEL 768
#define DINNER 1536
#define DSTATE 16
#define DTRANK 48
#define NPROJ  80      // DTRANK + 2*DSTATE
#define DIN2   3072    // 2*DINNER
#define LC     64      // scan chunk length
#define NCHUNK (LSEQ / LC)

typedef __bf16 bf16_t;
typedef __attribute__((ext_vector_type(16))) __bf16 v16bf;
typedef __attribute__((ext_vector_type(8)))  __bf16 v8bf;
typedef __attribute__((ext_vector_type(8)))  float  v8f;

// =====================================================================
// WMMA fragment loaders (contiguous k = 16*(lane/16) + s, same order in A and B).
// Work for both global and LDS source pointers (LDS -> ds_load_b128).
// =====================================================================

// A fragment, 16x32 (MxK) bf16. Lane l<16: row l, K=k0..k0+15;
// lane l>=16: row l-16, K=k0+16..k0+31.
__device__ __forceinline__ v16bf load_a_frag(const bf16_t* __restrict__ A,
                                             int lda, int m0, int k0, int lane) {
  const int r    = lane & 15;
  const int koff = (lane >> 4) << 4;            // 0 or 16
  return *reinterpret_cast<const v16bf*>(A + (size_t)(m0 + r) * lda + (k0 + koff));
}

// B fragment, 32x16 (KxN) bf16 where B[k][n] = W[n0+n][k].
// Lane l<16: column l, K=k0..k0+15; lane l>=16: column l-16, K=k0+16..k0+31.
__device__ __forceinline__ v16bf load_b_frag(const bf16_t* __restrict__ W,
                                             int ldb, int n0, int k0, int lane) {
  const int n    = lane & 15;
  const int koff = (lane >> 4) << 4;            // 0 or 16
  return *reinterpret_cast<const v16bf*>(W + (size_t)(n0 + n) * ldb + (k0 + koff));
}

#define GEMM_WAVES 4   // 128-thread blocks, 4 waves

// =====================================================================
// Direct-load WMMA GEMM (used for x_proj, N=80 needs per-wave guards).
// C[M,N] (f32) = A[M,K] (bf16 rm) * W[N,K]^T
// =====================================================================
template <int MT, int NT>
__global__ void __launch_bounds__(32 * GEMM_WAVES)
gemm_bf16_wmma(const bf16_t* __restrict__ A, const bf16_t* __restrict__ W,
               float* __restrict__ C, int M, int N, int K,
               int lda, int ldb, int ldc) {
  const int lane = threadIdx.x & 31;
  const int wave = threadIdx.x >> 5;
  const int m0 = blockIdx.y * (MT * 16);
  const int n0 = (blockIdx.x * GEMM_WAVES + wave) * (NT * 16);
  if (m0 >= M || n0 >= N) return;   // N is a multiple of 16

  v8f acc[MT][NT] = {};

  for (int k0 = 0; k0 < K; k0 += 32) {
    if (k0 + 32 < K) {  // WGP-scope prefetch of next K tile (global_prefetch_b8)
      __builtin_prefetch(A + (size_t)(m0 + (lane & 15)) * lda + k0 + 32, 0, 3);
      __builtin_prefetch(W + (size_t)(n0 + (lane & 15)) * ldb + k0 + 32, 0, 3);
    }
    v16bf af[MT];
    v16bf bfr[NT];
#pragma unroll
    for (int mi = 0; mi < MT; ++mi) af[mi] = load_a_frag(A, lda, m0 + 16 * mi, k0, lane);
#pragma unroll
    for (int ni = 0; ni < NT; ++ni) bfr[ni] = load_b_frag(W, ldb, n0 + 16 * ni, k0, lane);
#pragma unroll
    for (int mi = 0; mi < MT; ++mi)
#pragma unroll
      for (int ni = 0; ni < NT; ++ni)
        acc[mi][ni] = __builtin_amdgcn_wmma_f32_16x16x32_bf16(
            false, af[mi], false, bfr[ni], (short)0, acc[mi][ni], false, false);
  }

  const int cn   = lane & 15;
  const int rsel = (lane >> 4) << 3;  // D: VGPR j -> row j + 8*(lane>>4)
#pragma unroll
  for (int mi = 0; mi < MT; ++mi)
#pragma unroll
    for (int ni = 0; ni < NT; ++ni)
#pragma unroll
      for (int j = 0; j < 8; ++j)
        C[(size_t)(m0 + mi * 16 + rsel + j) * ldc + (n0 + ni * 16 + cn)] =
            acc[mi][ni][j];
}

// =====================================================================
// Async-LDS double-buffered WMMA GEMM (in_proj / out_proj).
// Block = 128 threads (4 waves) computes 64(M) x 256(N); per 32-K step
// each wave issues 10 global_load_async_to_lds_b128 (ASYNCcnt) for the
// 64x32 A tile + 256x32 B tile, overlapping the next tile's copies with
// the current tile's 16 WMMAs. Requires M%64==0, N%256==0, K%32==0.
// =====================================================================
#define KT_ELE 32
#define AT_ROWS 64
#define BT_ROWS 256

__device__ __forceinline__ void async_ld_b128(uint32_t lds_off, const bf16_t* g) {
  asm volatile("global_load_async_to_lds_b128 %0, %1, off"
               :: "v"(lds_off), "v"(g) : "memory");
}

__global__ void __launch_bounds__(128)
gemm_bf16_wmma_async(const bf16_t* __restrict__ A, const bf16_t* __restrict__ W,
                     float* __restrict__ C, int M, int N, int K,
                     int lda, int ldb, int ldc) {
  __shared__ bf16_t sA[2][AT_ROWS * KT_ELE];   //  8 KB
  __shared__ bf16_t sB[2][BT_ROWS * KT_ELE];   // 32 KB
  const int tid  = threadIdx.x;
  const int lane = tid & 31;
  const int wave = tid >> 5;
  const int mblk = blockIdx.y * AT_ROWS;
  const int nblk = blockIdx.x * BT_ROWS;
  const int n0w  = nblk + wave * 64;
  const int KT   = K / KT_ELE;

  auto issue_tile = [&](int kt, int buf) {
    const int k0 = kt * KT_ELE;
#pragma unroll
    for (int i = 0; i < 2; ++i) {              // A: 256 x 16B chunks
      const int ch = tid + i * 128;
      const int r = ch >> 2, s = ch & 3;
      async_ld_b128((uint32_t)(uintptr_t)&sA[buf][r * KT_ELE + s * 8],
                    A + (size_t)(mblk + r) * lda + k0 + s * 8);
    }
#pragma unroll
    for (int i = 0; i < 8; ++i) {              // B: 1024 x 16B chunks
      const int ch = tid + i * 128;
      const int r = ch >> 2, s = ch & 3;
      async_ld_b128((uint32_t)(uintptr_t)&sB[buf][r * KT_ELE + s * 8],
                    W + (size_t)(nblk + r) * ldb + k0 + s * 8);
    }
  };

  v8f acc[4][4] = {};
  issue_tile(0, 0);
  for (int kt = 0; kt < KT; ++kt) {
    const int buf = kt & 1;
    if (kt + 1 < KT) {
      issue_tile(kt + 1, buf ^ 1);             // prefetch next tile into other buffer
      asm volatile("s_wait_asynccnt 0xa" ::: "memory");   // tile kt copies done
    } else {
      asm volatile("s_wait_asynccnt 0x0" ::: "memory");
    }
    __syncthreads();                           // tile kt visible to all waves

    v16bf af[4], bfr[4];
#pragma unroll
    for (int mi = 0; mi < 4; ++mi)
      af[mi] = load_a_frag(&sA[buf][0], KT_ELE, mi * 16, 0, lane);
#pragma unroll
    for (int ni = 0; ni < 4; ++ni)
      bfr[ni] = load_b_frag(&sB[buf][0], KT_ELE, wave * 64 + ni * 16, 0, lane);
#pragma unroll
    for (int mi = 0; mi < 4; ++mi)
#pragma unroll
      for (int ni = 0; ni < 4; ++ni)
        acc[mi][ni] = __builtin_amdgcn_wmma_f32_16x16x32_bf16(
            false, af[mi], false, bfr[ni], (short)0, acc[mi][ni], false, false);
    __syncthreads();                           // reads done before buf is overwritten
  }

  const int cn   = lane & 15;
  const int rsel = (lane >> 4) << 3;
#pragma unroll
  for (int mi = 0; mi < 4; ++mi)
#pragma unroll
    for (int ni = 0; ni < 4; ++ni)
#pragma unroll
      for (int j = 0; j < 8; ++j)
        C[(size_t)(mblk + mi * 16 + rsel + j) * ldc + (n0w + ni * 16 + cn)] =
            acc[mi][ni][j];
}

// =====================================================================
// elementwise / pointwise kernels
// =====================================================================
__global__ void cvt_f32_bf16(const float* __restrict__ s, bf16_t* __restrict__ d, int n) {
  int i = blockIdx.x * blockDim.x + threadIdx.x;
  if (i < n) d[i] = (bf16_t)s[i];
}

// depthwise causal conv(4) + bias + SiLU; reads xs half of xz, writes f32 + bf16
__global__ void conv_silu(const float* __restrict__ xz, const float* __restrict__ w,
                          const float* __restrict__ bias,
                          float* __restrict__ xs_f, bf16_t* __restrict__ xs_h) {
  const int e = blockIdx.x * blockDim.x + threadIdx.x;
  const int l = blockIdx.y;
  const int b = blockIdx.z;
  float acc = bias[e];
#pragma unroll
  for (int j = 0; j < 4; ++j) {
    int li = l - 3 + j;
    if (li >= 0) acc += w[e * 4 + j] * xz[((size_t)(b * LSEQ + li)) * DIN2 + e];
  }
  float s = acc / (1.f + __expf(-acc));
  size_t o = ((size_t)(b * LSEQ + l)) * DINNER + e;
  xs_f[o] = s;
  xs_h[o] = (bf16_t)s;
}

// delta = softplus(dbc[:, :48] @ dtw^T + dtb); grid (DINNER/256, B*L)
__global__ void dt_softplus(const float* __restrict__ dbc, const float* __restrict__ dtw,
                            const float* __restrict__ dtb, float* __restrict__ delta) {
  __shared__ float sd[DTRANK];
  const int p   = blockIdx.y;
  const int tid = threadIdx.x;
  if (tid < DTRANK) sd[tid] = dbc[(size_t)p * NPROJ + tid];
  __syncthreads();
  const int e = blockIdx.x * blockDim.x + tid;
  float acc = dtb[e];
#pragma unroll 8
  for (int r = 0; r < DTRANK; ++r) acc += sd[r] * dtw[e * DTRANK + r];
  delta[(size_t)p * DINNER + e] = (acc > 20.f) ? acc : log1pf(__expf(acc));
}

// =====================================================================
// chunked selective scan (h_t = exp(delta*A) h_{t-1} + delta*xs*B)
// block = 512 threads: tid = n(16) x e_local(32); grid(E/32, NCHUNK, B)
// =====================================================================
__global__ void scan_pass1(const float* __restrict__ delta, const float* __restrict__ dbc,
                           const float* __restrict__ xs, const float* __restrict__ A_log,
                           float* __restrict__ hend, float* __restrict__ Pp, int dir) {
  const int tid = threadIdx.x;
  const int n  = tid & 15;
  const int e  = blockIdx.x * 32 + (tid >> 4);
  const int c  = blockIdx.y;
  const int b  = blockIdx.z;
  const float a = -__expf(A_log[e * DSTATE + n]);
  float h = 0.f, prod = 1.f;
  for (int i = 0; i < LC; ++i) {
    const int t = c * LC + i;
    const int l = dir ? (LSEQ - 1 - t) : t;
    const size_t base = (size_t)b * LSEQ + l;
    const float dv = delta[base * DINNER + e];
    const float xv = xs[base * DINNER + e];
    const float bv = dbc[base * NPROJ + DTRANK + n];
    const float dA = __expf(dv * a);
    h = dA * h + (dv * xv) * bv;
    prod *= dA;
  }
  const size_t idx = (((size_t)b * NCHUNK + c) * DINNER + e) * DSTATE + n;
  hend[idx] = h;
  Pp[idx]   = prod;
}

__global__ void scan_carry(const float* __restrict__ hend, const float* __restrict__ P,
                           float* __restrict__ hin) {
  const int t = blockIdx.x * blockDim.x + threadIdx.x;   // over B*E*N
  if (t >= BDIM * DINNER * DSTATE) return;
  const int b  = t / (DINNER * DSTATE);
  const int en = t % (DINNER * DSTATE);
  float carry = 0.f;
  for (int c = 0; c < NCHUNK; ++c) {
    const size_t idx = ((size_t)b * NCHUNK + c) * (DINNER * DSTATE) + en;
    hin[idx] = carry;
    carry = P[idx] * carry + hend[idx];
  }
}

__global__ void scan_pass2(const float* __restrict__ delta, const float* __restrict__ dbc,
                           const float* __restrict__ xs, const float* __restrict__ A_log,
                           const float* __restrict__ Dv, const float* __restrict__ hin,
                           float* __restrict__ y, int dir) {
  const int tid = threadIdx.x;
  const int n  = tid & 15;
  const int e  = blockIdx.x * 32 + (tid >> 4);
  const int c  = blockIdx.y;
  const int b  = blockIdx.z;
  const float a  = -__expf(A_log[e * DSTATE + n]);
  const float Dp = Dv[e];
  float h = hin[(((size_t)b * NCHUNK + c) * DINNER + e) * DSTATE + n];
  for (int i = 0; i < LC; ++i) {
    const int t = c * LC + i;
    const int l = dir ? (LSEQ - 1 - t) : t;
    const size_t base = (size_t)b * LSEQ + l;
    const float dv = delta[base * DINNER + e];
    const float xv = xs[base * DINNER + e];
    const float bv = dbc[base * NPROJ + DTRANK + n];
    const float cv = dbc[base * NPROJ + DTRANK + DSTATE + n];
    const float dA = __expf(dv * a);
    h = dA * h + (dv * xv) * bv;
    float contrib = h * cv;                    // reduce over n (16 lanes)
    contrib += __shfl_xor(contrib, 8, 16);
    contrib += __shfl_xor(contrib, 4, 16);
    contrib += __shfl_xor(contrib, 2, 16);
    contrib += __shfl_xor(contrib, 1, 16);
    if (n == 0) y[base * DINNER + e] = contrib + Dp * xv;
  }
}

// g = (y_f + y_b) * silu(z), to bf16 for the output GEMM
__global__ void gate_fuse(const float* __restrict__ yf, const float* __restrict__ yb,
                          const float* __restrict__ xz, bf16_t* __restrict__ g) {
  const size_t t = (size_t)blockIdx.x * blockDim.x + threadIdx.x;
  if (t >= (size_t)BDIM * LSEQ * DINNER) return;
  const size_t p = t / DINNER;
  const int    e = (int)(t % DINNER);
  const float zv = xz[p * DIN2 + DINNER + e];
  g[t] = (bf16_t)((yf[t] + yb[t]) * (zv / (1.f + __expf(-zv))));
}

// =====================================================================
static inline size_t align256(size_t x) { return (x + 255) & ~(size_t)255; }

extern "C" void kernel_launch(void* const* d_in, const int* in_sizes, int n_in,
                              void* d_out, int out_size, void* d_ws, size_t ws_size,
                              hipStream_t stream) {
  (void)in_sizes; (void)n_in; (void)out_size; (void)ws_size;
  const float* x       = (const float*)d_in[0];
  const float* w_in    = (const float*)d_in[1];
  const float* conv_w  = (const float*)d_in[2];
  const float* conv_b  = (const float*)d_in[3];
  const float* xpw_f   = (const float*)d_in[4];
  const float* dtw_f   = (const float*)d_in[5];
  const float* dtb_f   = (const float*)d_in[6];
  const float* A_log_f = (const float*)d_in[7];
  const float* D_f     = (const float*)d_in[8];
  const float* w_out   = (const float*)d_in[9];
  const float* A_log_b = (const float*)d_in[10];
  const float* xpw_b   = (const float*)d_in[11];
  const float* dtw_b   = (const float*)d_in[12];
  const float* dtb_b   = (const float*)d_in[13];
  const float* D_b     = (const float*)d_in[14];
  float* out = (float*)d_out;

  const int M = BDIM * LSEQ;      // 2048 token rows

  char* ws = (char*)d_ws;
  size_t off = 0;
  auto carve = [&](size_t bytes) -> void* { void* p = ws + off; off = align256(off + bytes); return p; };

  bf16_t* x_bf    = (bf16_t*)carve((size_t)M * DMODEL * 2);
  bf16_t* win_bf  = (bf16_t*)carve((size_t)DIN2 * DMODEL * 2);
  bf16_t* xpwf_bf = (bf16_t*)carve((size_t)NPROJ * DINNER * 2);
  bf16_t* xpwb_bf = (bf16_t*)carve((size_t)NPROJ * DINNER * 2);
  bf16_t* wout_bf = (bf16_t*)carve((size_t)DMODEL * DINNER * 2);
  float*  xz      = (float*)carve((size_t)M * DIN2 * 4);
  float*  xs_f    = (float*)carve((size_t)M * DINNER * 4);
  bf16_t* xs_h    = (bf16_t*)carve((size_t)M * DINNER * 2);
  float*  dbc_f   = (float*)carve((size_t)M * NPROJ * 4);
  float*  dbc_b   = (float*)carve((size_t)M * NPROJ * 4);
  float*  delta_f = (float*)carve((size_t)M * DINNER * 4);
  float*  delta_b = (float*)carve((size_t)M * DINNER * 4);
  float*  hend    = (float*)carve((size_t)BDIM * NCHUNK * DINNER * DSTATE * 4);
  float*  Pbuf    = (float*)carve((size_t)BDIM * NCHUNK * DINNER * DSTATE * 4);
  float*  hin     = (float*)carve((size_t)BDIM * NCHUNK * DINNER * DSTATE * 4);
  float*  y_f     = (float*)carve((size_t)M * DINNER * 4);
  float*  y_b     = (float*)carve((size_t)M * DINNER * 4);
  bf16_t* g_bf    = (bf16_t*)carve((size_t)M * DINNER * 2);

  // --- bf16 conversions for WMMA operands ---
  auto cvt = [&](const float* s, bf16_t* d, int n) {
    cvt_f32_bf16<<<dim3((n + 255) / 256), dim3(256), 0, stream>>>(s, d, n);
  };
  cvt(x,     x_bf,    M * DMODEL);
  cvt(w_in,  win_bf,  DIN2 * DMODEL);
  cvt(xpw_f, xpwf_bf, NPROJ * DINNER);
  cvt(xpw_b, xpwb_bf, NPROJ * DINNER);
  cvt(w_out, wout_bf, DMODEL * DINNER);

  // --- in_proj: xz (2048 x 3072), async-LDS WMMA path ---
  gemm_bf16_wmma_async<<<dim3(DIN2 / BT_ROWS, M / AT_ROWS), dim3(128), 0, stream>>>(
      x_bf, win_bf, xz, M, DIN2, DMODEL, DMODEL, DMODEL, DIN2);

  // --- conv + SiLU ---
  conv_silu<<<dim3(DINNER / 256, LSEQ, BDIM), dim3(256), 0, stream>>>(
      xz, conv_w, conv_b, xs_f, xs_h);

  // --- x_proj (both directions), N=80: direct-load WMMA path ---
  gemm_bf16_wmma<4, 1><<<dim3(2, M / 64), dim3(128), 0, stream>>>(
      xs_h, xpwf_bf, dbc_f, M, NPROJ, DINNER, DINNER, DINNER, NPROJ);
  gemm_bf16_wmma<4, 1><<<dim3(2, M / 64), dim3(128), 0, stream>>>(
      xs_h, xpwb_bf, dbc_b, M, NPROJ, DINNER, DINNER, DINNER, NPROJ);

  // --- dt_proj + softplus ---
  dt_softplus<<<dim3(DINNER / 256, M), dim3(256), 0, stream>>>(dbc_f, dtw_f, dtb_f, delta_f);
  dt_softplus<<<dim3(DINNER / 256, M), dim3(256), 0, stream>>>(dbc_b, dtw_b, dtb_b, delta_b);

  // --- selective scans (chunked two-pass linear recurrence) ---
  const dim3 sgrid(DINNER / 32, NCHUNK, BDIM), sblk(512);
  const dim3 cgrid((BDIM * DINNER * DSTATE + 255) / 256), cblk(256);
  scan_pass1<<<sgrid, sblk, 0, stream>>>(delta_f, dbc_f, xs_f, A_log_f, hend, Pbuf, 0);
  scan_carry<<<cgrid, cblk, 0, stream>>>(hend, Pbuf, hin);
  scan_pass2<<<sgrid, sblk, 0, stream>>>(delta_f, dbc_f, xs_f, A_log_f, D_f, hin, y_f, 0);
  scan_pass1<<<sgrid, sblk, 0, stream>>>(delta_b, dbc_b, xs_f, A_log_b, hend, Pbuf, 1);
  scan_carry<<<cgrid, cblk, 0, stream>>>(hend, Pbuf, hin);
  scan_pass2<<<sgrid, sblk, 0, stream>>>(delta_b, dbc_b, xs_f, A_log_b, D_b, hin, y_b, 1);

  // --- gate + out_proj (async-LDS WMMA path) ---
  gate_fuse<<<dim3((unsigned)(((size_t)M * DINNER + 255) / 256)), dim3(256), 0, stream>>>(
      y_f, y_b, xz, g_bf);
  gemm_bf16_wmma_async<<<dim3(DMODEL / BT_ROWS, M / AT_ROWS), dim3(128), 0, stream>>>(
      g_bf, wout_bf, out, M, DMODEL, DINNER, DINNER, DINNER, DMODEL);
}
